// GPT_24386824307242
// MI455X (gfx1250) — matrix operand order repaired and checked by reading, not compiled, not run on hardware
//
#include <hip/hip_runtime.h>
#include <math.h>

// GPT-2 forward (B=2,T=1024,E=1024,H=16,L=6,FF=4096,V=50257) for gfx1250.
// - All GEMMs: v_wmma_f32_16x16x32_f16, f32 accumulate, 128x128 block tile,
//   8 waves (wave32), each wave 32x64 -> 8 WMMA per K-step, reuse-A hints.
// - f16 operands end-to-end; weights converted fp32->f16 *transposed* so
//   every weight GEMM takes the TRANS_B path: both A and B tiles are
//   contiguous per lane and staged with GLOBAL_LOAD_ASYNC_TO_LDS_B128
//   (ASYNCcnt) + s_wait_asynccnt.
// - Double-buffered LDS ping-pong: async-stage tile k+1 while WMMAs consume
//   tile k; a single s_wait_asynccnt + s_barrier per K-step.
// - Causality exploited structurally: fully-masked score blocks skipped,
//   P@V K-loop clipped at the block's last valid key.

#define E_DIM   1024
#define H_DIM   16
#define HD_DIM  64
#define L_NUM   6
#define B_DIM   2
#define T_DIM   1024
#define FF_DIM  4096
#define V_DIM   50257
#define M_ROWS  (B_DIM * T_DIM)   // 2048

typedef _Float16 v16h __attribute__((ext_vector_type(16)));
typedef _Float16 v8h  __attribute__((ext_vector_type(8)));
typedef _Float16 v2h  __attribute__((ext_vector_type(2)));
typedef float    v8f  __attribute__((ext_vector_type(8)));

__device__ __forceinline__ float gelu_exact(float v) {
    return 0.5f * v * (1.0f + erff(v * 0.70710678118654752f));
}

// ---- CDNA5 async global->LDS copy helpers (ASYNCcnt) ----------------------
__device__ __forceinline__ unsigned lds_off32(const void* p) {
    // LDS aperture: flat addr low 32 bits are the LDS byte offset.
    return (unsigned)(unsigned long long)p;
}
__device__ __forceinline__ void async_ld_b128(unsigned lds_addr,
                                              const _Float16* gaddr) {
    asm volatile("global_load_async_to_lds_b128 %0, %1, off"
                 :: "v"(lds_addr), "v"(gaddr)
                 : "memory");
}
__device__ __forceinline__ void async_wait0() {
    asm volatile("s_wait_asynccnt 0x0" ::: "memory");
}

// WMMA with literal reuse-A flag (must be a frontend constant).
#define WMMA_F16(a, b, c, ra) \
    __builtin_amdgcn_wmma_f32_16x16x32_f16(false, (a), false, (b), \
                                           (short)0, (c), (ra), false)

// ---------------------------------------------------------------------------
// LDS-tiled WMMA GEMM, f16 in / f32 accum, double-buffered async staging:
//   acc = A[z] @ B[z];  v = act(alpha*acc + bias) + resid
//   Cf (fp32, optional) and/or Ch (f16, optional) get v.
// Block tile 128(M) x 128(N), K-step 32, 256 threads = 8 waves.
// Wave w: rows (w>>1)*32, cols (w&1)*64 -> 2x4 grid of 16x16 WMMA tiles.
// A: f16 [M,K] row-major (lda).  B: TRANS_B=1 -> element (k,n) at
// Bm[n*ldb+k] (async path); TRANS_B=0 -> f16 [K,N] (ldb, scatter staging).
// LDS keeps Bs[n][k].  z offsets: off = (z/zdiv)*s?0 + (z%zdiv)*s?1.
// causal: 0=none, 1=skip fully-masked blocks (n0>m0+127, scores GEMM),
//         2=clip K at m0+128 (P@V GEMM).
// ---------------------------------------------------------------------------
template <bool TRANS_B>
__global__ __launch_bounds__(256) void wmma_gemm(
    const _Float16* __restrict__ A,  long lda, long sA0, long sA1,
    const _Float16* __restrict__ Bm, long ldb, long sB0, long sB1,
    float*          __restrict__ Cf,
    _Float16*       __restrict__ Ch, long ldc, long sC0, long sC1,
    const float* __restrict__ bias,
    const float* __restrict__ resid,
    int M, int N, int K, float alpha, int act, int zdiv, int causal)
{
    __shared__ _Float16 As[2][128 * 32];   // [m][k]
    __shared__ _Float16 Bs[2][128 * 32];   // [n][k]

    const int m0 = blockIdx.y * 128;
    const int n0 = blockIdx.x * 128;

    // causal==1: output block entirely above the diagonal -> fully masked;
    // downstream softmax never reads cols > q, so skip it (block-uniform).
    if (causal == 1 && n0 > m0 + 127) return;
    // causal==2: P[q,k]==0 for k>q -> clip the K loop at this block's
    // last valid key (m0+128 is a multiple of the K-step).
    const int Keff = (causal == 2 && m0 + 128 < K) ? m0 + 128 : K;

    const int tid  = threadIdx.x;
    const int z    = blockIdx.z;
    const int zq   = z / zdiv;
    const int zr   = z - zq * zdiv;
    const long offA = (long)zq * sA0 + (long)zr * sA1;
    const long offB = (long)zq * sB0 + (long)zr * sB1;
    const long offC = (long)zq * sC0 + (long)zr * sC1;

    const int lane = tid & 31;
    const int wave = tid >> 5;
    const int r    = lane & 15;
    const int hi   = lane >> 4;
    const int wm   = (wave >> 1) * 32;   // wave row base in block tile
    const int wn   = (wave & 1)  * 64;   // wave col base in block tile

    // staging coordinates (A, and B in TRANS layout): 16 halfs per thread
    const int srow = tid >> 1;           // 0..127
    const int skof = (tid & 1) * 16;     // 0 or 16

    // stage K-tile k0 into buffer `buf` (async for contiguous paths)
    auto stage = [&](int buf, int k0) {
        {   // A tile 128x32, contiguous 32B per thread
            const _Float16* ag =
                A + offA + (long)(m0 + srow) * lda + (k0 + skof);
            unsigned la = lds_off32(&As[buf][srow * 32 + skof]);
            async_ld_b128(la,      ag);
            async_ld_b128(la + 16, ag + 8);
        }
        if (TRANS_B) {
            const int gn = n0 + srow;
            _Float16* bs = &Bs[buf][srow * 32 + skof];
            if (gn < N) {
                const _Float16* bg =
                    Bm + offB + (long)gn * ldb + (k0 + skof);
                unsigned lb = lds_off32(bs);
                async_ld_b128(lb,      bg);
                async_ld_b128(lb + 16, bg + 8);
            } else {
                v2h zz = {};
#pragma unroll
                for (int i = 0; i < 8; ++i) *(v2h*)(bs + 2 * i) = zz;
            }
        } else {
            // transpose during staging: 16 halfs along n at fixed k
            const int bk = tid >> 3;             // 0..31
            const int bn = (tid & 7) * 16;       // 0..112
            const _Float16* bg =
                Bm + offB + (long)(k0 + bk) * ldb + (n0 + bn);
            if (n0 + 128 <= N) {
                v8h q0 = *(const v8h*)bg;
                v8h q1 = *(const v8h*)(bg + 8);
#pragma unroll
                for (int i = 0; i < 8; ++i) {
                    Bs[buf][(bn + i) * 32 + bk]     = q0[i];
                    Bs[buf][(bn + 8 + i) * 32 + bk] = q1[i];
                }
            } else {
#pragma unroll
                for (int i = 0; i < 16; ++i)
                    Bs[buf][(bn + i) * 32 + bk] =
                        (n0 + bn + i < N) ? bg[i] : (_Float16)0.0f;
            }
        }
    };

    v8f acc[2][4] = {};

    stage(0, 0);
    async_wait0();
    __syncthreads();

    int cur = 0;
    for (int k0 = 0; k0 < Keff; k0 += 32) {
        const int nxt = cur ^ 1;
        if (k0 + 32 < Keff) stage(nxt, k0 + 32);  // overlap with compute

        // ---- fragments (ISA 7.12.2 16-bit layouts) ----
        v16h af[2], bf[4];
#pragma unroll
        for (int v = 0; v < 8; ++v) {
            const int kk = (v < 4 ? 2 * v : 8 + 2 * v) + hi * 8;
#pragma unroll
            for (int mi = 0; mi < 2; ++mi) {
                v2h p = *(const v2h*)(&As[cur][(wm + mi * 16 + r) * 32 + kk]);
                af[mi][2 * v] = p[0]; af[mi][2 * v + 1] = p[1];
            }
        }
#pragma unroll
        for (int v = 0; v < 8; ++v) {
            const int kk = (v < 4 ? 2 * v : 8 + 2 * v) + hi * 8;
#pragma unroll
            for (int ni = 0; ni < 4; ++ni) {
                v2h p = *(const v2h*)(&Bs[cur][(wn + ni * 16 + r) * 32 + kk]);
                bf[ni][2 * v] = p[0]; bf[ni][2 * v + 1] = p[1];
            }
        }
        // reuse_a hint (ISA 7.12 OPSEL RA): next WMMA in issue order
        // consumes the same A fragment for the first 3 of each group.
        acc[0][0] = WMMA_F16(af[0], bf[0], acc[0][0], true);
        acc[0][1] = WMMA_F16(af[0], bf[1], acc[0][1], true);
        acc[0][2] = WMMA_F16(af[0], bf[2], acc[0][2], true);
        acc[0][3] = WMMA_F16(af[0], bf[3], acc[0][3], false);
        acc[1][0] = WMMA_F16(af[1], bf[0], acc[1][0], true);
        acc[1][1] = WMMA_F16(af[1], bf[1], acc[1][1], true);
        acc[1][2] = WMMA_F16(af[1], bf[2], acc[1][2], true);
        acc[1][3] = WMMA_F16(af[1], bf[3], acc[1][3], false);

        async_wait0();        // our async stage of `nxt` landed in LDS
        __syncthreads();      // everyone staged `nxt`, everyone consumed `cur`
        cur = nxt;
    }

    // ---- epilogue: lane (l&15)=col, VGPR p = row (+8 for hi) ----
#pragma unroll
    for (int mi = 0; mi < 2; ++mi)
#pragma unroll
        for (int ni = 0; ni < 4; ++ni) {
            const int col = n0 + wn + ni * 16 + r;
            if (col < N) {
#pragma unroll
                for (int p = 0; p < 8; ++p) {
                    const int row = m0 + wm + mi * 16 + hi * 8 + p;
                    float v = acc[mi][ni][p] * alpha;
                    if (bias) v += bias[col];
                    if (act == 1) v = gelu_exact(v);
                    const long ci = offC + (long)row * ldc + col;
                    if (resid) v += resid[ci];
                    if (Cf) Cf[ci] = v;
                    if (Ch) Ch[ci] = (_Float16)v;
                }
            }
        }
}

// ---------------------------------------------------------------------------
// x[row,:] = wte[idx[row],:] + wpe[row%T,:]   (fp32 residual stream)
// ---------------------------------------------------------------------------
__global__ __launch_bounds__(256) void embed_kernel(
    const int* __restrict__ idx, const float* __restrict__ wte,
    const float* __restrict__ wpe, float* __restrict__ x)
{
    const int row = blockIdx.x;
    const int t   = row % T_DIM;
    const int tok = idx[row];
    const int c   = threadIdx.x * 4;
    float4 a = *(const float4*)(wte + (long)tok * E_DIM + c);
    float4 p = *(const float4*)(wpe + (long)t * E_DIM + c);
    float4 o = {a.x + p.x, a.y + p.y, a.z + p.z, a.w + p.w};
    *(float4*)(x + (long)row * E_DIM + c) = o;
}

// ---------------------------------------------------------------------------
// fp32 -> f16 plain copy (wte for the tied head). n multiple of 1024.
// ---------------------------------------------------------------------------
__global__ __launch_bounds__(256) void cvt_f32_f16(
    const float* __restrict__ in, _Float16* __restrict__ out)
{
    const long i = (long)blockIdx.x * 1024 + (long)threadIdx.x * 4;
    float4 q = *(const float4*)(in + i);
    v2h a = {(_Float16)q.x, (_Float16)q.y};
    v2h b = {(_Float16)q.z, (_Float16)q.w};
    *(v2h*)(out + i)     = a;
    *(v2h*)(out + i + 2) = b;
}

// ---------------------------------------------------------------------------
// fp32 [K,N] -> f16 transposed [N,K] (weights), 32x32 tiles via LDS.
// Block (32,8); grid (N/32, K/32). Coalesced on both sides.
// ---------------------------------------------------------------------------
__global__ __launch_bounds__(256) void cvtT_f32_f16(
    const float* __restrict__ in, _Float16* __restrict__ out, int K, int N)
{
    __shared__ _Float16 t[32][33];
    const int nb = blockIdx.x * 32;
    const int kb = blockIdx.y * 32;
    const int tx = threadIdx.x;          // 0..31
    const int ty = threadIdx.y;          // 0..7
#pragma unroll
    for (int i = 0; i < 32; i += 8)
        t[ty + i][tx] = (_Float16)in[(long)(kb + ty + i) * N + (nb + tx)];
    __syncthreads();
#pragma unroll
    for (int i = 0; i < 32; i += 8)
        out[(long)(nb + ty + i) * K + (kb + tx)] = t[tx][ty + i];
}

// ---------------------------------------------------------------------------
// LayerNorm over E=1024 (one block, 128 threads = 4 wave32, per row), f16 out.
// ---------------------------------------------------------------------------
__global__ __launch_bounds__(128) void ln_kernel(
    const float* __restrict__ in, const float* __restrict__ g,
    const float* __restrict__ b, _Float16* __restrict__ out)
{
    __shared__ float red[4];
    const int row = blockIdx.x;
    const int tid = threadIdx.x;
    const float* xr = in + (long)row * E_DIM;

    float v[8];
    float s = 0.0f;
#pragma unroll
    for (int i = 0; i < 8; ++i) { v[i] = xr[tid + i * 128]; s += v[i]; }
#pragma unroll
    for (int off = 16; off; off >>= 1) s += __shfl_xor(s, off, 32);
    if ((tid & 31) == 0) red[tid >> 5] = s;
    __syncthreads();
    const float mean = (red[0] + red[1] + red[2] + red[3]) * (1.0f / E_DIM);

    float vs = 0.0f;
#pragma unroll
    for (int i = 0; i < 8; ++i) { float d = v[i] - mean; vs += d * d; }
#pragma unroll
    for (int off = 16; off; off >>= 1) vs += __shfl_xor(vs, off, 32);
    __syncthreads();
    if ((tid & 31) == 0) red[tid >> 5] = vs;
    __syncthreads();
    const float var  = (red[0] + red[1] + red[2] + red[3]) * (1.0f / E_DIM);
    const float rstd = rsqrtf(var + 1e-5f);
#pragma unroll
    for (int i = 0; i < 8; ++i) {
        const int c = tid + i * 128;
        out[(long)row * E_DIM + c] =
            (_Float16)((v[i] - mean) * rstd * g[c] + b[c]);
    }
}

// ---------------------------------------------------------------------------
// Causal softmax on f16 scores, in place. Row S[z, q, :T], valid cols 0..q.
// ---------------------------------------------------------------------------
__global__ __launch_bounds__(128) void softmax_causal(_Float16* __restrict__ S)
{
    __shared__ float red[4];
    const int q   = blockIdx.x;
    const int z   = blockIdx.y;
    const int tid = threadIdx.x;
    _Float16* row = S + ((long)z * T_DIM + q) * T_DIM;

    float v[8];
    float mx = -3.4e38f;
#pragma unroll
    for (int i = 0; i < 8; ++i) {
        const int c = tid + i * 128;
        v[i] = (c <= q) ? (float)row[c] : -3.4e38f;
        mx = fmaxf(mx, v[i]);
    }
#pragma unroll
    for (int off = 16; off; off >>= 1) mx = fmaxf(mx, __shfl_xor(mx, off, 32));
    if ((tid & 31) == 0) red[tid >> 5] = mx;
    __syncthreads();
    mx = fmaxf(fmaxf(red[0], red[1]), fmaxf(red[2], red[3]));

    float s = 0.0f;
#pragma unroll
    for (int i = 0; i < 8; ++i) {
        const int c = tid + i * 128;
        v[i] = (c <= q) ? expf(v[i] - mx) : 0.0f;
        s += v[i];
    }
#pragma unroll
    for (int off = 16; off; off >>= 1) s += __shfl_xor(s, off, 32);
    __syncthreads();
    if ((tid & 31) == 0) red[tid >> 5] = s;
    __syncthreads();
    const float inv = 1.0f / (red[0] + red[1] + red[2] + red[3]);
#pragma unroll
    for (int i = 0; i < 8; ++i)
        row[tid + i * 128] = (_Float16)(v[i] * inv);
}

// ---------------------------------------------------------------------------
extern "C" void kernel_launch(void* const* d_in, const int* in_sizes, int n_in,
                              void* d_out, int out_size, void* d_ws, size_t ws_size,
                              hipStream_t stream)
{
    (void)in_sizes; (void)n_in; (void)out_size; (void)ws_size;
    const int*   idx   = (const int*)  d_in[0];
    const float* wte   = (const float*)d_in[1];
    const float* wpe   = (const float*)d_in[2];
    const float* ln1_g = (const float*)d_in[3];
    const float* ln1_b = (const float*)d_in[4];
    const float* wqkv  = (const float*)d_in[5];
    const float* bqkv  = (const float*)d_in[6];
    const float* wproj = (const float*)d_in[7];
    const float* bproj = (const float*)d_in[8];
    const float* ln2_g = (const float*)d_in[9];
    const float* ln2_b = (const float*)d_in[10];
    const float* wfc   = (const float*)d_in[11];
    const float* bfc   = (const float*)d_in[12];
    const float* wout  = (const float*)d_in[13];
    const float* bout  = (const float*)d_in[14];
    const float* lnf_g = (const float*)d_in[15];
    const float* lnf_b = (const float*)d_in[16];
    float* out = (float*)d_out;

    // workspace: x fp32 residual stream, then f16 region
    float*    x     = (float*)d_ws;                              // [2048,1024] f32
    _Float16* f16b  = (_Float16*)(x + (size_t)M_ROWS * E_DIM);
    _Float16* h_h   = f16b;                                      // [2048,1024]
    _Float16* qkv_h = h_h   + (size_t)M_ROWS * E_DIM;            // [2048,3072]
    _Float16* S_h   = qkv_h + (size_t)M_ROWS * 3 * E_DIM;        // [32,1024,1024]
    _Float16* y_h   = S_h   + (size_t)B_DIM * H_DIM * T_DIM * T_DIM; // [2048,1024]
    _Float16* hff_h = y_h   + (size_t)M_ROWS * E_DIM;            // [2048,4096]
    _Float16* wb    = hff_h + (size_t)M_ROWS * FF_DIM;           // <= [4096,1024] (transposed weights)
    _Float16* wte_h = wb    + (size_t)E_DIM * FF_DIM;            // [50257,1024]

    const long zQ  = (long)T_DIM * 3 * E_DIM;   // batch stride in qkv_h
    const long zTT = (long)T_DIM * T_DIM;       // head stride in S_h

    embed_kernel<<<M_ROWS, 256, 0, stream>>>(idx, wte, wpe, x);

    const dim3 tb(32, 8);
    for (int l = 0; l < L_NUM; ++l) {
        ln_kernel<<<M_ROWS, 128, 0, stream>>>(x, ln1_g + l * E_DIM,
                                              ln1_b + l * E_DIM, h_h);
        // qkv_h = f16(h @ Wqkv + bqkv)      [2048,1024]@[1024,3072]
        cvtT_f32_f16<<<dim3(3 * E_DIM / 32, E_DIM / 32), tb, 0, stream>>>(
            wqkv + (size_t)l * E_DIM * 3 * E_DIM, wb, E_DIM, 3 * E_DIM);
        wmma_gemm<true><<<dim3(3 * E_DIM / 128, M_ROWS / 128, 1), 256, 0, stream>>>(
            h_h, E_DIM, 0, 0,
            wb, E_DIM, 0, 0,
            nullptr, qkv_h, 3 * E_DIM, 0, 0,
            bqkv + l * 3 * E_DIM, nullptr,
            M_ROWS, 3 * E_DIM, E_DIM, 1.0f, 0, 1, 0);
        // S_h[z] = f16((q @ k^T)/8)         per (b,h): [1024,64]@[64,1024]
        // causal=1: skip fully-masked blocks above the diagonal.
        wmma_gemm<true><<<dim3(T_DIM / 128, T_DIM / 128, B_DIM * H_DIM), 256, 0, stream>>>(
            qkv_h,         3 * E_DIM, zQ, HD_DIM,
            qkv_h + E_DIM, 3 * E_DIM, zQ, HD_DIM,
            nullptr, S_h, T_DIM, (long)H_DIM * zTT, zTT,
            nullptr, nullptr,
            T_DIM, T_DIM, HD_DIM, 0.125f, 0, H_DIM, 1);
        softmax_causal<<<dim3(T_DIM, B_DIM * H_DIM), 128, 0, stream>>>(S_h);
        // y_h[z] = f16(P @ V)               per (b,h): [1024,1024]@[1024,64]
        // causal=2: clip K at the block's last valid key.
        wmma_gemm<false><<<dim3(1, T_DIM / 128, B_DIM * H_DIM), 256, 0, stream>>>(
            S_h, T_DIM, (long)H_DIM * zTT, zTT,
            qkv_h + 2 * E_DIM, 3 * E_DIM, zQ, HD_DIM,
            nullptr, y_h, E_DIM, (long)T_DIM * E_DIM, HD_DIM,
            nullptr, nullptr,
            T_DIM, HD_DIM, T_DIM, 1.0f, 0, H_DIM, 2);
        // x += y @ Wproj + bproj
        cvtT_f32_f16<<<dim3(E_DIM / 32, E_DIM / 32), tb, 0, stream>>>(
            wproj + (size_t)l * E_DIM * E_DIM, wb, E_DIM, E_DIM);
        wmma_gemm<true><<<dim3(E_DIM / 128, M_ROWS / 128, 1), 256, 0, stream>>>(
            y_h, E_DIM, 0, 0,
            wb, E_DIM, 0, 0,
            x, nullptr, E_DIM, 0, 0,
            bproj + l * E_DIM, x,
            M_ROWS, E_DIM, E_DIM, 1.0f, 0, 1, 0);
        ln_kernel<<<M_ROWS, 128, 0, stream>>>(x, ln2_g + l * E_DIM,
                                              ln2_b + l * E_DIM, h_h);
        // hff_h = f16(gelu(h @ Wfc + bfc))  [2048,1024]@[1024,4096]
        cvtT_f32_f16<<<dim3(FF_DIM / 32, E_DIM / 32), tb, 0, stream>>>(
            wfc + (size_t)l * E_DIM * FF_DIM, wb, E_DIM, FF_DIM);
        wmma_gemm<true><<<dim3(FF_DIM / 128, M_ROWS / 128, 1), 256, 0, stream>>>(
            h_h, E_DIM, 0, 0,
            wb, E_DIM, 0, 0,
            nullptr, hff_h, FF_DIM, 0, 0,
            bfc + l * FF_DIM, nullptr,
            M_ROWS, FF_DIM, E_DIM, 1.0f, 1, 1, 0);
        // x += hff @ Wout + bout            [2048,4096]@[4096,1024]
        cvtT_f32_f16<<<dim3(E_DIM / 32, FF_DIM / 32), tb, 0, stream>>>(
            wout + (size_t)l * FF_DIM * E_DIM, wb, FF_DIM, E_DIM);
        wmma_gemm<true><<<dim3(E_DIM / 128, M_ROWS / 128, 1), 256, 0, stream>>>(
            hff_h, FF_DIM, 0, 0,
            wb, FF_DIM, 0, 0,
            x, nullptr, E_DIM, 0, 0,
            bout + l * E_DIM, x,
            M_ROWS, E_DIM, FF_DIM, 1.0f, 0, 1, 0);
    }

    ln_kernel<<<M_ROWS, 128, 0, stream>>>(x, lnf_g, lnf_b, h_h);
    // logits = h @ wte^T   (wte [V,E] is already the transposed layout)
    cvt_f32_f16<<<V_DIM, 256, 0, stream>>>(wte, wte_h);  // V*E/1024 = V blocks
    wmma_gemm<true><<<dim3((V_DIM + 127) / 128, M_ROWS / 128, 1), 256, 0, stream>>>(
        h_h, E_DIM, 0, 0,
        wte_h, E_DIM, 0, 0,
        out, nullptr, V_DIM, 0, 0,
        nullptr, nullptr,
        M_ROWS, V_DIM, E_DIM, 1.0f, 0, 1, 0);
}
